// TransformerModel_15994458211448
// MI455X (gfx1250) — matrix operand order repaired and checked by reading, not compiled, not run on hardware
//
#include <hip/hip_runtime.h>
#include <hip/hip_bf16.h>

// ---------------------------------------------------------------------------
// TransformerConv GNN on gfx1250 (wave32): TDM tensor_load_to_lds for B tiles,
// WMMA bf16 16x16x32 for all dense math, f32 atomics for segment softmax.
//
// Sizes fixed by the reference: N=10000, E=320000, H=4, C=64, IN=256, ED=32,
// OUT=2.  All M are multiples of 16, all K of 32, all N of 32 -> no tails.
// ---------------------------------------------------------------------------

#define NN   10000
#define EE   320000
#define HH   4
#define CC   64
#define HC   256      // H*C
#define ED_  32
#define OUTD 2

typedef __bf16 v16bf __attribute__((ext_vector_type(16)));
typedef float  v8f   __attribute__((ext_vector_type(8)));
typedef unsigned u32x4 __attribute__((ext_vector_type(4)));
typedef int      i32x8 __attribute__((ext_vector_type(8)));
typedef int      i32x4 __attribute__((ext_vector_type(4)));

// ---- wave32 helpers --------------------------------------------------------
__device__ __forceinline__ float wave_sum(float v) {
#pragma unroll
  for (int o = 16; o > 0; o >>= 1) v += __shfl_xor(v, o, 32);
  return v;
}
// monotone float<->uint encoding so unsigned atomicMax == float max
__device__ __forceinline__ unsigned encf(float f) {
  int i = __float_as_int(f);
  return (i >= 0) ? ((unsigned)i ^ 0x80000000u) : ~(unsigned)i;
}
__device__ __forceinline__ float decf(unsigned u) {
  int i = (u & 0x80000000u) ? (int)(u ^ 0x80000000u) : (int)~u;
  return __int_as_float(i);
}

// ---------------------------------------------------------------------------
// TDM: DMA a 2-D fp32 tile [rows x cols] (row stride in elements) from global
// memory into LDS at byte offset lds_off, contiguous.  Descriptor per ISA
// 8.3/8.4: group0 {count=1, lds_addr, global_addr, type=2}; group1
// {data_size=4B, tensor_dim0/1, tile_dim0/1, tensor_dim0_stride}; groups 2,3
// zero (<=2-D).  Wave-level op (descriptors -> SGPRs, EXEC ignored): call it
// from ONE wave only, then s_wait_tensorcnt + workgroup barrier.
// This toolchain exposes the 6-arg builtin:
//   (uint32x4, int32x8, int32x4, int32x4, int32x8, i32 cpol)
// ---------------------------------------------------------------------------
__device__ __forceinline__ void tdm_load_tile_f32(unsigned lds_off,
                                                  const float* gptr,
                                                  int rows, int cols,
                                                  int row_stride) {
  const unsigned long long ga = (unsigned long long)(uintptr_t)gptr;
  u32x4 g0;
  g0[0] = 1u;                                    // count=1 (valid user D#)
  g0[1] = lds_off;                               // lds_addr (bytes)
  g0[2] = (unsigned)(ga & 0xFFFFFFFFu);          // global_addr[31:0]
  g0[3] = (unsigned)((ga >> 32) & 0x1FFFFFFu)    // global_addr[56:32]
          | (2u << 30);                          // type=2 ("image")
  i32x8 g1;
  g1[0] = 0x20000;                               // data_size=2 (4B), mask=0
  g1[1] = (cols & 0xFFFF) << 16;                 // tensor_dim0[15:0]
  g1[2] = (rows & 0xFFFF) << 16;                 // tensor_dim1[15:0]
  g1[3] = (cols & 0xFFFF) << 16;                 // tile_dim0
  g1[4] = (rows & 0xFFFF);                       // tile_dim1 (tile_dim2=0)
  g1[5] = row_stride;                            // tensor_dim0_stride[31:0]
  g1[6] = 0;
  g1[7] = 0;
  i32x4 gz4 = {0, 0, 0, 0};
  i32x8 gz8 = {0, 0, 0, 0, 0, 0, 0, 0};
  __builtin_amdgcn_tensor_load_to_lds(g0, g1, gz4, gz4, gz8, 0);
}

// ---------------------------------------------------------------------------
// WMMA GEMM: C[M,N] = A[M,K] @ B[K,N] (+bias).  fp32 in memory, bf16
// fragments, f32 accumulation.  BT=true: B stored transposed Bt[N,K] (ldb).
// Block = 256 threads = 8 waves -> 128 rows x 32 cols of C; each wave owns a
// 16x32 strip (two 16x16 accumulators, A fragment reused).  Per k-step the
// 32x32 fp32 B tile is TDM-DMA'd into LDS once and shared by all waves.
// Fragment packing per ISA 7.12.2 (bf16 A/B/C wave32 layouts).
// ---------------------------------------------------------------------------
template <bool BT>
__global__ void wmma_gemm_kernel(const float* __restrict__ A,
                                 const float* __restrict__ B,
                                 const float* __restrict__ bias,
                                 float* __restrict__ C,
                                 int M, int K, int N,
                                 int lda, int ldb, int ldc) {
  __shared__ float sB[32 * 32];                  // one 32(K) x 32(N) tile
  const int wave = threadIdx.x >> 5;
  const int lane = threadIdx.x & 31;
  int m0 = (blockIdx.x * 8 + wave) * 16;
  if (m0 + 16 > M) m0 = M - 16;                  // clamp: keep all waves in barriers
  const int n0 = blockIdx.y * 32;

  const int r  = lane & 15;                      // A row in tile / B,C column
  const int hi = lane >> 4;                      // K-half select
  const unsigned sB_off = (unsigned)(uintptr_t)(void*)sB;

  v8f acc0 = {}, acc1 = {};
  const float* arow = A + (size_t)(m0 + r) * lda;

  for (int k0 = 0; k0 < K; k0 += 32) {
    __syncthreads();                             // LDS tile free for reuse
    if (wave == 0) {
      // BT: tile rows = out-columns, contiguous dim = K  -> LDS [n][k]
      // !BT: tile rows = K, contiguous dim = out-columns -> LDS [k][n]
      const float* gp = BT ? (B + (size_t)n0 * ldb + k0)
                           : (B + (size_t)k0 * ldb + n0);
      tdm_load_tile_f32(sB_off, gp, 32, 32, ldb);
      __builtin_amdgcn_s_wait_tensorcnt(0);
    }
    __syncthreads();                             // publish tile

    if (k0 + 32 < K) __builtin_prefetch(arow + k0 + 32, 0, 1);

    v16bf af;
#pragma unroll
    for (int j = 0; j < 16; ++j) {               // A: rows r, K {0..7,16..23}+8*hi
      const int ka = ((j < 8) ? j : j + 8) + hi * 8;
      af[j] = (__bf16)arow[k0 + ka];
    }
#pragma unroll
    for (int s = 0; s < 2; ++s) {                // two 16-col subtiles
      const int col = r + 16 * s;
      v16bf bf_;
#pragma unroll
      for (int j = 0; j < 16; ++j) {             // B: col, K {0..7,16..23}+8*hi
        const int kk = ((j < 8) ? j : j + 8) + hi * 8;
        bf_[j] = (__bf16)(BT ? sB[col * 32 + kk] : sB[kk * 32 + col]);
      }
      if (s == 0)
        acc0 = __builtin_amdgcn_wmma_f32_16x16x32_bf16(
            false, af, false, bf_, (short)0, acc0, false, false);
      else
        acc1 = __builtin_amdgcn_wmma_f32_16x16x32_bf16(
            false, af, false, bf_, (short)0, acc1, false, false);
    }
  }

  const int rowb = m0 + hi * 8;                  // C layout: vgpr i -> row rowb+i
#pragma unroll
  for (int s = 0; s < 2; ++s) {
    const int col = n0 + r + 16 * s;
    const float bb = bias ? bias[col] : 0.0f;
    const v8f& acc = s ? acc1 : acc0;
#pragma unroll
    for (int i = 0; i < 8; ++i)
      C[(size_t)(rowb + i) * ldc + col] = acc[i] + bb;
  }
}

// ---------------------------------------------------------------------------
// Edge pass 1: alpha[e,h] = ( q[d,h]·k[s,h] + t[d,h]·ea[e] ) / sqrt(C)
// one wave per edge; segment max via encoded atomicMax.
// t[n, h*32+l] = sum_c q[n,h*64+c] * We[l, h*64+c]  (precomputed GEMM)
// ---------------------------------------------------------------------------
__global__ void edge_alpha_kernel(const long long* __restrict__ ei,
                                  const float* __restrict__ ea,
                                  const float* __restrict__ q,
                                  const float* __restrict__ k,
                                  const float* __restrict__ t,
                                  float* __restrict__ alpha,
                                  unsigned* __restrict__ amax) {
  const int wave = threadIdx.x >> 5, lane = threadIdx.x & 31;
  const int e = blockIdx.x * 8 + wave;
  if (e >= EE) return;
  const int s = (int)ei[e];
  const int d = (int)ei[EE + e];
  const float eal = ea[(size_t)e * ED_ + lane];
#pragma unroll
  for (int h = 0; h < HH; ++h) {
    const float* qp = q + (size_t)d * HC + h * CC;
    const float* kp = k + (size_t)s * HC + h * CC;
    float p = qp[2 * lane] * kp[2 * lane] + qp[2 * lane + 1] * kp[2 * lane + 1];
    p += t[(size_t)d * (HH * ED_) + h * ED_ + lane] * eal;
    p = wave_sum(p);
    if (lane == 0) {
      const float al = p * 0.125f;               // 1/sqrt(64)
      alpha[(size_t)e * HH + h] = al;
      atomicMax(&amax[d * HH + h], encf(al));
    }
  }
}

// Edge pass 2: ex = exp(alpha - amax[dst]); den[dst] += ex
__global__ void edge_exp_kernel(const long long* __restrict__ ei,
                                const unsigned* __restrict__ amax,
                                float* __restrict__ alpha,
                                float* __restrict__ den) {
  const int id = blockIdx.x * blockDim.x + threadIdx.x;
  if (id >= EE * HH) return;
  const int e = id >> 2, h = id & 3;
  const int d = (int)ei[EE + e];
  const float ex = __expf(alpha[id] - decf(amax[d * HH + h]));
  alpha[id] = ex;
  atomicAdd(&den[d * HH + h], ex);
}

// Edge pass 3: w = ex/den[dst];  u[dst,h,:] += w*ea[e,:];
//              outv[dst,h,:] += w*v[src,h,:]   (one wave per edge)
__global__ void edge_scatter_kernel(const long long* __restrict__ ei,
                                    const float* __restrict__ ea,
                                    const float* __restrict__ v,
                                    const float* __restrict__ alpha,
                                    const float* __restrict__ den,
                                    float* __restrict__ u,
                                    float* __restrict__ outv) {
  const int wave = threadIdx.x >> 5, lane = threadIdx.x & 31;
  const int e = blockIdx.x * 8 + wave;
  if (e >= EE) return;
  const int s = (int)ei[e];
  const int d = (int)ei[EE + e];
  const float eal = ea[(size_t)e * ED_ + lane];
#pragma unroll
  for (int h = 0; h < HH; ++h) {
    const float w = alpha[(size_t)e * HH + h] / den[d * HH + h];
    atomicAdd(&u[(size_t)d * (HH * ED_) + h * ED_ + lane], w * eal);
    const float* vp = v + (size_t)s * HC + h * CC;
    float* op = outv + (size_t)d * HC + h * CC;
    atomicAdd(&op[2 * lane],     w * vp[2 * lane]);
    atomicAdd(&op[2 * lane + 1], w * vp[2 * lane + 1]);
  }
}

// h[n,c] = mean_h( outv[n,h,c] + eo[n,h,c] ) + skip[n,c]
__global__ void combine_kernel(const float* __restrict__ outv,
                               const float* __restrict__ eo,
                               const float* __restrict__ skip,
                               float* __restrict__ hout) {
  const int id = blockIdx.x * blockDim.x + threadIdx.x;
  if (id >= NN * CC) return;
  const int n = id >> 6, c = id & 63;
  float s = 0.f;
#pragma unroll
  for (int h = 0; h < HH; ++h)
    s += outv[(size_t)n * HC + h * CC + c] + eo[(size_t)n * HC + h * CC + c];
  hout[id] = 0.25f * s + skip[id];
}

// LayerNorm over 64 channels (+optional ReLU); one wave per row, 2 elems/lane
__global__ void layernorm_kernel(const float* __restrict__ x,
                                 const float* __restrict__ w,
                                 const float* __restrict__ b,
                                 float* __restrict__ y, int relu) {
  const int wave = threadIdx.x >> 5, lane = threadIdx.x & 31;
  const int row = blockIdx.x * 8 + wave;
  if (row >= NN) return;
  const float x0 = x[(size_t)row * CC + lane];
  const float x1 = x[(size_t)row * CC + 32 + lane];
  const float mu = wave_sum(x0 + x1) * (1.0f / 64.0f);
  const float d0 = x0 - mu, d1 = x1 - mu;
  const float var = wave_sum(d0 * d0 + d1 * d1) * (1.0f / 64.0f);
  const float rs = rsqrtf(var + 1e-5f);
  float y0 = d0 * rs * w[lane] + b[lane];
  float y1 = d1 * rs * w[32 + lane] + b[32 + lane];
  if (relu) { y0 = fmaxf(y0, 0.f); y1 = fmaxf(y1, 0.f); }
  y[(size_t)row * CC + lane]      = y0;
  y[(size_t)row * CC + 32 + lane] = y1;
}

// out[n,:] = h[n,:] @ Wf[64,2] + bf
__global__ void final_kernel(const float* __restrict__ h,
                             const float* __restrict__ Wf,
                             const float* __restrict__ bfv,
                             float* __restrict__ out) {
  const int n = blockIdx.x * blockDim.x + threadIdx.x;
  if (n >= NN) return;
  float o0 = bfv[0], o1 = bfv[1];
#pragma unroll
  for (int c = 0; c < CC; ++c) {
    const float hv = h[(size_t)n * CC + c];
    o0 += hv * Wf[c * 2];
    o1 += hv * Wf[c * 2 + 1];
  }
  out[n * 2] = o0;
  out[n * 2 + 1] = o1;
}

// ---------------------------------------------------------------------------
// host side
// ---------------------------------------------------------------------------
struct Scratch {
  float *q, *k, *v, *skip, *t, *alpha, *den, *u, *outv, *eo, *hA, *hB, *hC;
  unsigned* amax;
};

static void run_layer(const float* x, int din,
                      const float* Wq, const float* bq,
                      const float* Wk, const float* bk,
                      const float* Wv, const float* bv,
                      const float* We, const float* Ws, const float* bs,
                      const long long* ei, const float* ea,
                      const Scratch& S, float* hout, hipStream_t st) {
  const dim3 blk(256);
  (void)hipMemsetAsync(S.amax, 0, (size_t)NN * HH * sizeof(unsigned), st); // enc(-inf)
  (void)hipMemsetAsync(S.den,  0, (size_t)NN * HH * sizeof(float), st);
  (void)hipMemsetAsync(S.u,    0, (size_t)NN * HH * ED_ * sizeof(float), st);
  (void)hipMemsetAsync(S.outv, 0, (size_t)NN * HC * sizeof(float), st);

  const int gm = (NN + 127) / 128;
  // Q,K,V,skip projections
  wmma_gemm_kernel<false><<<dim3(gm, HC / 32), blk, 0, st>>>(x, Wq, bq, S.q, NN, din, HC, din, HC, HC);
  wmma_gemm_kernel<false><<<dim3(gm, HC / 32), blk, 0, st>>>(x, Wk, bk, S.k, NN, din, HC, din, HC, HC);
  wmma_gemm_kernel<false><<<dim3(gm, HC / 32), blk, 0, st>>>(x, Wv, bv, S.v, NN, din, HC, din, HC, HC);
  wmma_gemm_kernel<false><<<dim3(gm, CC / 32), blk, 0, st>>>(x, Ws, bs, S.skip, NN, din, CC, din, CC, CC);
  // t[:,h,:] = q_h @ We_h^T   (pre-reduces q·eproj to a 32-dot per edge)
  for (int h = 0; h < HH; ++h)
    wmma_gemm_kernel<true><<<dim3(gm, ED_ / 32), blk, 0, st>>>(
        S.q + h * CC, We + h * CC, nullptr, S.t + h * ED_,
        NN, CC, ED_, HC, HC, HH * ED_);
  // edge phase (segment softmax + scatter)
  edge_alpha_kernel<<<(EE + 7) / 8, blk, 0, st>>>(ei, ea, S.q, S.k, S.t, S.alpha, S.amax);
  edge_exp_kernel<<<(EE * HH + 255) / 256, blk, 0, st>>>(ei, S.amax, S.alpha, S.den);
  edge_scatter_kernel<<<(EE + 7) / 8, blk, 0, st>>>(ei, ea, S.v, S.alpha, S.den, S.u, S.outv);
  // eo[:,h,:] = u_h @ We_h   (fused edge-feature contribution, WMMA)
  for (int h = 0; h < HH; ++h)
    wmma_gemm_kernel<false><<<dim3(gm, CC / 32), blk, 0, st>>>(
        S.u + h * ED_, We + h * CC, nullptr, S.eo + h * CC,
        NN, ED_, CC, HH * ED_, HC, HC);
  combine_kernel<<<(NN * CC + 255) / 256, blk, 0, st>>>(S.outv, S.eo, S.skip, hout);
}

extern "C" void kernel_launch(void* const* d_in, const int* in_sizes, int n_in,
                              void* d_out, int out_size, void* d_ws, size_t ws_size,
                              hipStream_t stream) {
  // inputs in setup_inputs() order (params tuples flattened)
  const float*     x   = (const float*)d_in[0];
  const long long* ei  = (const long long*)d_in[1];   // int64 [2,E]
  const float*     ea  = (const float*)d_in[2];
  const float *Wq1 = (const float*)d_in[3],  *bq1 = (const float*)d_in[4];
  const float *Wk1 = (const float*)d_in[5],  *bk1 = (const float*)d_in[6];
  const float *Wv1 = (const float*)d_in[7],  *bv1 = (const float*)d_in[8];
  const float *We1 = (const float*)d_in[9];
  const float *Ws1 = (const float*)d_in[10], *bs1 = (const float*)d_in[11];
  const float *Wq2 = (const float*)d_in[12], *bq2 = (const float*)d_in[13];
  const float *Wk2 = (const float*)d_in[14], *bk2 = (const float*)d_in[15];
  const float *Wv2 = (const float*)d_in[16], *bv2 = (const float*)d_in[17];
  const float *We2 = (const float*)d_in[18];
  const float *Ws2 = (const float*)d_in[19], *bs2 = (const float*)d_in[20];
  const float *lnw = (const float*)d_in[21], *lnb = (const float*)d_in[22];
  const float *Wf  = (const float*)d_in[23], *bfv = (const float*)d_in[24];
  float* out = (float*)d_out;
  (void)in_sizes; (void)n_in; (void)out_size; (void)ws_size;

  // carve workspace (~75 MB of fp32)
  float* w = (float*)d_ws;
  Scratch S;
  S.q = w;     w += (size_t)NN * HC;
  S.k = w;     w += (size_t)NN * HC;
  S.v = w;     w += (size_t)NN * HC;
  S.skip = w;  w += (size_t)NN * CC;
  S.t = w;     w += (size_t)NN * HH * ED_;
  S.alpha = w; w += (size_t)EE * HH;
  S.den = w;   w += (size_t)NN * HH;
  S.u = w;     w += (size_t)NN * HH * ED_;
  S.outv = w;  w += (size_t)NN * HC;
  S.eo = w;    w += (size_t)NN * HC;
  S.hA = w;    w += (size_t)NN * CC;
  S.hB = w;    w += (size_t)NN * CC;
  S.hC = w;    w += (size_t)NN * CC;
  S.amax = (unsigned*)w;

  const dim3 blk(256);
  // layer 1 (din=256) -> hA, LN+ReLU -> hB
  run_layer(x, 256, Wq1, bq1, Wk1, bk1, Wv1, bv1, We1, Ws1, bs1, ei, ea, S, S.hA, stream);
  layernorm_kernel<<<(NN + 7) / 8, blk, 0, stream>>>(S.hA, lnw, lnb, S.hB, 1);
  // layer 2 (din=64) -> hA, LN -> hC
  run_layer(S.hB, 64, Wq2, bq2, Wk2, bk2, Wv2, bv2, We2, Ws2, bs2, ei, ea, S, S.hA, stream);
  layernorm_kernel<<<(NN + 7) / 8, blk, 0, stream>>>(S.hA, lnw, lnb, S.hC, 0);
  // readout
  final_kernel<<<(NN + 255) / 256, blk, 0, stream>>>(S.hC, Wf, bfv, out);
}
